// DeepseekV32Indexer_80848464380580
// MI455X (gfx1250) — compile-verified, matrix operand order checked by reading, not run on hardware
//
#include <hip/hip_runtime.h>
#include <hip/hip_bf16.h>

// ---------------------------------------------------------------------------
// DeepSeek-V3.2 indexer for MI455X (gfx1250, wave32, WMMA).
// All heavy GEMMs run on v_wmma_f32_16x16x32_bf16 (bf16 in, f32 accumulate).
// ---------------------------------------------------------------------------

#define S_LEN    2048
#define D_DIM    7168
#define R_DIM    1536
#define NHEADS   64
#define HDIM     128
#define QWIDTH   (NHEADS * HDIM)     // 8192
#define KW_N     (HDIM + NHEADS)     // 192 : [k | w] fused GEMM width
#define ROPE_HALF 32                 // ROPE_DIM/2
#define TOPK     1024
#define INV_SQRT_HD 0.08838834764831845f   // 128^-0.5 (also Hadamard scale)

typedef __attribute__((ext_vector_type(16))) __bf16 v16bf;
typedef __attribute__((ext_vector_type(8)))  __bf16 v8bf;
typedef __attribute__((ext_vector_type(8)))  float  v8f;

static __device__ __forceinline__ __bf16 f32_to_bf16(float f) {
    unsigned u = __float_as_uint(f);
    unsigned r = (u + 0x7FFFu + ((u >> 16) & 1u)) >> 16;
    unsigned short h = (unsigned short)r;
    __bf16 b;
    __builtin_memcpy(&b, &h, 2);
    return b;
}

// Load one 16x32 A-fragment (or 32x16 B-fragment) row for this lane from a
// row-major bf16 [row][K] array.  Per the CDNA5 ISA layout, lane (l15 = lane&15)
// holds row/col l15; half = lane>>4 selects K sub-ranges:
//   elements 0..7  <- K = half*8 .. half*8+7        (contiguous 16 bytes)
//   elements 8..15 <- K = 16+half*8 .. 16+half*8+7  (contiguous 16 bytes)
static __device__ __forceinline__ v16bf load_frag(const __bf16* row_k, int half) {
    const v8bf lo = *(const v8bf*)(row_k + half * 8);
    const v8bf hi = *(const v8bf*)(row_k + half * 8 + 16);
    v16bf f;
#pragma unroll
    for (int i = 0; i < 8; ++i) { f[i] = lo[i]; f[i + 8] = hi[i]; }
    return f;
}

static __device__ __forceinline__ v8f wmma_bf16(v16bf a, v16bf b, v8f c) {
    return __builtin_amdgcn_wmma_f32_16x16x32_bf16(
        /*neg_a=*/false, a, /*neg_b=*/false, b,
        /*c_mod=*/(short)0, c, /*reuse_a=*/false, /*reuse_b=*/false);
}

// ---------------------------------------------------------------------------
// 1) elementwise fp32 -> bf16
// ---------------------------------------------------------------------------
__global__ void f32_to_bf16_kernel(const float* __restrict__ in,
                                   __bf16* __restrict__ out, int n) {
    int i = blockIdx.x * 256 + threadIdx.x;
    if (i < n) out[i] = f32_to_bf16(in[i]);
}

// ---------------------------------------------------------------------------
// 2) transpose + convert: in[R][C] fp32 -> out[C][R] bf16 (row-major, K-minor)
// ---------------------------------------------------------------------------
__global__ void transpose_conv_kernel(const float* __restrict__ in,
                                      __bf16* __restrict__ out, int R, int C) {
    __shared__ float tile[32][33];
    const int c0 = blockIdx.x * 32, r0 = blockIdx.y * 32;
    const int tx = threadIdx.x & 31, ty = threadIdx.x >> 5;  // 256 thr: ty 0..7
    for (int i = ty; i < 32; i += 8) {
        int r = r0 + i, c = c0 + tx;
        tile[i][tx] = (r < R && c < C) ? in[(long)r * C + c] : 0.f;
    }
    __syncthreads();
    for (int i = ty; i < 32; i += 8) {
        int c = c0 + i, r = r0 + tx;
        if (c < C && r < R) out[(long)c * R + r] = f32_to_bf16(tile[tx][i]);
    }
}

// ---------------------------------------------------------------------------
// 3) generic WMMA GEMM: C[M][N] f32 = A[M][K]bf16 @ B[N][K]bf16^T
//    block = 256 threads (8 waves), tile 32(M) x 64(N), one 16x16 tile/wave.
//    Requires M%32==0, N%64==0, K%32==0.
// ---------------------------------------------------------------------------
__global__ void gemm_bf16_nt_kernel(const __bf16* __restrict__ A,
                                    const __bf16* __restrict__ B,
                                    float* __restrict__ C,
                                    int M, int N, int K) {
    const int lane = threadIdx.x & 31, wid = threadIdx.x >> 5;
    const int half = lane >> 4, l15 = lane & 15;
    const int m_frag = blockIdx.x * 32 + (wid & 1) * 16 + l15;
    const int n_frag = blockIdx.y * 64 + (wid >> 1) * 16 + l15;
    const __bf16* arow = A + (long)m_frag * K;
    const __bf16* brow = B + (long)n_frag * K;
    v8f acc = {};
    for (int kb = 0; kb < K; kb += 32) {
        v16bf af = load_frag(arow + kb, half);
        v16bf bf_ = load_frag(brow + kb, half);
        acc = wmma_bf16(af, bf_, acc);
    }
    // D layout: lane l, vgpr v -> row = v + 8*(lane>>4), col = lane&15
    const int orow = blockIdx.x * 32 + (wid & 1) * 16 + 8 * half;
    const int ocol = blockIdx.y * 64 + (wid >> 1) * 16 + l15;
#pragma unroll
    for (int v = 0; v < 8; ++v)
        C[(long)(orow + v) * N + ocol] = acc[v];
}

// ---------------------------------------------------------------------------
// 4) K/w epilogue: per row s of kw_out[S][192]:
//    k = FWHT(rope(layernorm(kw[:,0:128]))) * 128^-0.5  -> bf16
//    w = kw[:,128:192] * NHEADS^-0.5                    -> f32
//    block = 128 threads, one row.
// ---------------------------------------------------------------------------
__global__ void kw_epilogue_kernel(const float* __restrict__ kw,
                                   const float* __restrict__ cosb,
                                   const float* __restrict__ sinb,
                                   const float* __restrict__ knw,
                                   const float* __restrict__ knb,
                                   __bf16* __restrict__ Kout,
                                   float* __restrict__ wout) {
    __shared__ float row[128];
    __shared__ float red[128];
    const int s = blockIdx.x, t = threadIdx.x;
    const float x = kw[(long)s * KW_N + t];
    red[t] = x;
    __syncthreads();
    for (int o = 64; o > 0; o >>= 1) {
        if (t < o) red[t] += red[t + o];
        __syncthreads();
    }
    const float mu = red[0] * (1.f / 128.f);
    __syncthreads();
    const float d = x - mu;
    red[t] = d * d;
    __syncthreads();
    for (int o = 64; o > 0; o >>= 1) {
        if (t < o) red[t] += red[t + o];
        __syncthreads();
    }
    const float var = red[0] * (1.f / 128.f);
    row[t] = d * rsqrtf(var + 1e-6f) * knw[t] + knb[t];
    __syncthreads();
    if (t < 32) {  // interleaved RoPE on dims [0,64)
        float xr = row[2 * t], xi = row[2 * t + 1];
        float co = cosb[s * ROPE_HALF + t], si = sinb[s * ROPE_HALF + t];
        row[2 * t]     = xr * co - xi * si;
        row[2 * t + 1] = xr * si + xi * co;
    }
    __syncthreads();
    for (int h = 1; h < 128; h <<= 1) {  // FWHT == x @ Sylvester-H (unnormalized)
        if (t < 64) {
            int i = (t / h) * 2 * h + (t % h);
            float a = row[i], b = row[i + h];
            row[i] = a + b;
            row[i + h] = a - b;
        }
        __syncthreads();
    }
    Kout[(long)s * HDIM + t] = f32_to_bf16(row[t] * INV_SQRT_HD);
    if (t < NHEADS) wout[(long)s * NHEADS + t] = kw[(long)s * KW_N + HDIM + t] * 0.125f;
}

// ---------------------------------------------------------------------------
// 5) Q projection, fused rope + FWHT.  grid = (S/32, NHEADS), 256 threads.
//    Wave wid: m strip = (wid&1)*16, n strip = (wid>>1)*32 (two 16x16 tiles).
// ---------------------------------------------------------------------------
__global__ void q_proj_kernel(const __bf16* __restrict__ qlora,   // [S][R]
                              const __bf16* __restrict__ wqbT,    // [QWIDTH][R]
                              const float* __restrict__ cosb,
                              const float* __restrict__ sinb,
                              __bf16* __restrict__ Qout) {        // [S][QWIDTH]
    __shared__ float qt[32][129];
    const int tid = threadIdx.x, lane = tid & 31, wid = tid >> 5;
    const int half = lane >> 4, l15 = lane & 15;
    const int m0 = blockIdx.x * 32;
    const int h  = blockIdx.y;
    const int ms = (wid & 1) * 16, ns = (wid >> 1) * 32;

    const __bf16* arow = qlora + (long)(m0 + ms + l15) * R_DIM;
    const __bf16* b0   = wqbT + (long)(h * HDIM + ns + l15) * R_DIM;
    const __bf16* b1   = b0 + (long)16 * R_DIM;
    v8f acc0 = {}, acc1 = {};
    for (int kb = 0; kb < R_DIM; kb += 32) {
        v16bf a  = load_frag(arow + kb, half);
        v16bf f0 = load_frag(b0 + kb, half);
        v16bf f1 = load_frag(b1 + kb, half);
        acc0 = wmma_bf16(a, f0, acc0);
        acc1 = wmma_bf16(a, f1, acc1);
    }
    const int rb = ms + 8 * half;
#pragma unroll
    for (int v = 0; v < 8; ++v) {
        qt[rb + v][ns + l15]      = acc0[v];
        qt[rb + v][ns + 16 + l15] = acc1[v];
    }
    __syncthreads();

    const int r  = tid >> 3;          // 0..31 : local row
    const int c0 = (tid & 7) * 16;    // 16-col chunk
    const int s  = m0 + r;
    if (c0 < 64) {                    // RoPE on dims [0,64)
#pragma unroll
        for (int j = 0; j < 8; ++j) {
            int dd = (c0 >> 1) + j;
            float xr = qt[r][c0 + 2 * j], xi = qt[r][c0 + 2 * j + 1];
            float co = cosb[s * ROPE_HALF + dd], si = sinb[s * ROPE_HALF + dd];
            qt[r][c0 + 2 * j]     = xr * co - xi * si;
            qt[r][c0 + 2 * j + 1] = xr * si + xi * co;
        }
    }
    __syncthreads();
    for (int hstep = 1; hstep < 128; hstep <<= 1) {   // FWHT, 7 stages
        const int pbase = (tid & 7) * 8;
#pragma unroll
        for (int j = 0; j < 8; ++j) {
            int p = pbase + j;
            int i = (p / hstep) * 2 * hstep + (p % hstep);
            float a = qt[r][i], b = qt[r][i + hstep];
            qt[r][i] = a + b;
            qt[r][i + hstep] = a - b;
        }
        __syncthreads();
    }
    __bf16* orow = Qout + (long)s * QWIDTH + h * HDIM;
#pragma unroll
    for (int j = 0; j < 16; ++j)
        orow[c0 + j] = f32_to_bf16(qt[r][c0 + j] * INV_SQRT_HD);
}

// ---------------------------------------------------------------------------
// 6) Scores: 64x64 (s,t) tile per block, loop over 64 heads.
//    scores[s,t] = sum_h relu(q[s,h,:]*k[t,:]) * 128^-0.5 * w[s,h], causal.
// ---------------------------------------------------------------------------
#define KT_STRIDE 136   // padded row stride (bf16) to avoid LDS bank conflicts

__global__ void scores_kernel(const __bf16* __restrict__ Q,     // [S][QWIDTH]
                              const __bf16* __restrict__ Kmat,  // [S][HDIM]
                              const float* __restrict__ w,      // [S][NHEADS]
                              float* __restrict__ scores) {     // [S][S]
    const int t0 = blockIdx.x * 64, s0 = blockIdx.y * 64;
    const int tid = threadIdx.x, lane = tid & 31, wid = tid >> 5;
    const int half = lane >> 4, l15 = lane & 15;
    const float NEG_INF = -__builtin_inff();

    if (t0 > s0) {  // whole tile above diagonal -> -inf
        const int r = tid >> 2, cb = (tid & 3) * 16;
#pragma unroll
        for (int j = 0; j < 16; ++j)
            scores[(long)(s0 + r) * S_LEN + t0 + cb + j] = NEG_INF;
        return;
    }

    __shared__ __bf16 kt[64 * KT_STRIDE];
    __shared__ float  wt[64 * NHEADS];
    {
        const int r = tid >> 2, cb = (tid & 3) * 32;
        const __bf16* src = Kmat + (long)(t0 + r) * HDIM + cb;
        v8bf* dst = (v8bf*)(kt + r * KT_STRIDE + cb);
#pragma unroll
        for (int j = 0; j < 4; ++j) dst[j] = *(const v8bf*)(src + j * 8);
        const int cw = (tid & 3) * 16;
        const float* wsrc = w + (long)(s0 + r) * NHEADS + cw;
#pragma unroll
        for (int j = 0; j < 16; ++j) wt[r * NHEADS + cw + j] = wsrc[j];
    }
    __syncthreads();

    const int ss = (wid & 3) * 16;   // s strip (16 rows)
    const int ts = (wid >> 2) * 32;  // t strip (two 16-col tiles)
    v8f acc0 = {}, acc1 = {};
    const __bf16* qrow = Q + (long)(s0 + ss + l15) * QWIDTH;
    const __bf16* k0 = kt + (ts + l15) * KT_STRIDE;
    const __bf16* k1 = kt + (ts + 16 + l15) * KT_STRIDE;

    for (int h = 0; h < NHEADS; ++h) {
        const __bf16* qh = qrow + h * HDIM;
        __builtin_prefetch((const void*)(qh + HDIM), 0, 1);  // next head's Q row
        v8f d0 = {}, d1 = {};
#pragma unroll
        for (int kb = 0; kb < HDIM; kb += 32) {
            v16bf a = load_frag(qh + kb, half);
            d0 = wmma_bf16(a, load_frag(k0 + kb, half), d0);
            d1 = wmma_bf16(a, load_frag(k1 + kb, half), d1);
        }
#pragma unroll
        for (int v = 0; v < 8; ++v) {
            const float wv = wt[(ss + 8 * half + v) * NHEADS + h] * INV_SQRT_HD;
            float r0 = d0[v]; r0 = r0 > 0.f ? r0 : 0.f;
            float r1 = d1[v]; r1 = r1 > 0.f ? r1 : 0.f;
            acc0[v] += r0 * wv;
            acc1[v] += r1 * wv;
        }
    }
    const int srow = s0 + ss + 8 * half;
    const int tc0 = t0 + ts + l15, tc1 = tc0 + 16;
#pragma unroll
    for (int v = 0; v < 8; ++v) {
        const int s = srow + v;
        scores[(long)s * S_LEN + tc0] = (tc0 <= s) ? acc0[v] : NEG_INF;
        scores[(long)s * S_LEN + tc1] = (tc1 <= s) ? acc1[v] : NEG_INF;
    }
}

// ---------------------------------------------------------------------------
// 7) Top-k: per-row bitonic sort of 2048 (value, index) pairs; jax tie-break
//    (descending value, ascending index).  Writes first 1024 indices.
// ---------------------------------------------------------------------------
__global__ void topk_kernel(const float* __restrict__ scores,
                            int* __restrict__ topk) {
    __shared__ float v[S_LEN];
    __shared__ int   ix[S_LEN];
    const int s = blockIdx.x, tid = threadIdx.x;
    for (int i = tid; i < S_LEN; i += 256) {
        v[i] = scores[(long)s * S_LEN + i];
        ix[i] = i;
    }
    __syncthreads();
    for (int k = 2; k <= S_LEN; k <<= 1) {
        for (int j = k >> 1; j > 0; j >>= 1) {
            for (int i = tid; i < S_LEN; i += 256) {
                const int p = i ^ j;
                if (p > i) {
                    const bool up = ((i & k) == 0);
                    const float va = v[i], vb = v[p];
                    const int ia = ix[i], ib = ix[p];
                    const bool prec_pi = (vb > va) || (vb == va && ib < ia);
                    const bool prec_ip = (va > vb) || (va == vb && ia < ib);
                    if (up ? prec_pi : prec_ip) {
                        v[i] = vb; v[p] = va;
                        ix[i] = ib; ix[p] = ia;
                    }
                }
            }
            __syncthreads();
        }
    }
    for (int i = tid; i < TOPK; i += 256)
        topk[(long)s * TOPK + i] = ix[i];
}

// ---------------------------------------------------------------------------
// launcher
// ---------------------------------------------------------------------------
extern "C" void kernel_launch(void* const* d_in, const int* in_sizes, int n_in,
                              void* d_out, int out_size, void* d_ws, size_t ws_size,
                              hipStream_t stream) {
    (void)in_sizes; (void)n_in; (void)out_size; (void)ws_size;
    const float* hidden = (const float*)d_in[0];   // [1,2048,7168]
    const float* qlora  = (const float*)d_in[1];   // [1,2048,1536]
    const float* cosb   = (const float*)d_in[2];   // [2048,32]
    const float* sinb   = (const float*)d_in[3];   // [2048,32]
    const float* wqb    = (const float*)d_in[4];   // [1536,8192]
    const float* wkw    = (const float*)d_in[5];   // [7168,128]
    const float* knw    = (const float*)d_in[6];   // [128]
    const float* knb    = (const float*)d_in[7];   // [128]
    const float* wproj  = (const float*)d_in[8];   // [7168,64]

    float* scores = (float*)d_out;                              // 2048*2048 f32
    int*   topk   = (int*)(scores + (size_t)S_LEN * S_LEN);     // 2048*1024 i32

    char* ws = (char*)d_ws;
    auto alloc = [&](size_t bytes) {
        char* p = ws;
        ws += (bytes + 255) & ~(size_t)255;
        return p;
    };
    __bf16* hidden_bf = (__bf16*)alloc((size_t)S_LEN * D_DIM * 2);
    __bf16* wcatT     = (__bf16*)alloc((size_t)KW_N * D_DIM * 2);   // [192][7168]
    float*  kw_out    = (float*) alloc((size_t)S_LEN * KW_N * 4);
    __bf16* Kbf       = (__bf16*)alloc((size_t)S_LEN * HDIM * 2);
    float*  wmat      = (float*) alloc((size_t)S_LEN * NHEADS * 4);
    __bf16* qlora_bf  = (__bf16*)alloc((size_t)S_LEN * R_DIM * 2);
    __bf16* wqbT      = (__bf16*)alloc((size_t)QWIDTH * R_DIM * 2); // [8192][1536]
    __bf16* Qbf       = (__bf16*)alloc((size_t)S_LEN * QWIDTH * 2);

    // --- converts / transposes ---
    {
        int n = S_LEN * D_DIM;
        f32_to_bf16_kernel<<<(n + 255) / 256, 256, 0, stream>>>(hidden, hidden_bf, n);
    }
    transpose_conv_kernel<<<dim3(HDIM / 32, D_DIM / 32), 256, 0, stream>>>(
        wkw, wcatT, D_DIM, HDIM);                       // rows 0..127 of wcatT
    transpose_conv_kernel<<<dim3(NHEADS / 32, D_DIM / 32), 256, 0, stream>>>(
        wproj, wcatT + (size_t)HDIM * D_DIM, D_DIM, NHEADS);  // rows 128..191
    {
        int n = S_LEN * R_DIM;
        f32_to_bf16_kernel<<<(n + 255) / 256, 256, 0, stream>>>(qlora, qlora_bf, n);
    }
    transpose_conv_kernel<<<dim3(QWIDTH / 32, R_DIM / 32), 256, 0, stream>>>(
        wqb, wqbT, R_DIM, QWIDTH);

    // --- k/w fused GEMM + epilogue ---
    gemm_bf16_nt_kernel<<<dim3(S_LEN / 32, KW_N / 64), 256, 0, stream>>>(
        hidden_bf, wcatT, kw_out, S_LEN, KW_N, D_DIM);
    kw_epilogue_kernel<<<S_LEN, 128, 0, stream>>>(
        kw_out, cosb, sinb, knw, knb, Kbf, wmat);

    // --- Q projection (GEMM + rope + FWHT fused) ---
    q_proj_kernel<<<dim3(S_LEN / 32, NHEADS), 256, 0, stream>>>(
        qlora_bf, wqbT, cosb, sinb, Qbf);

    // --- scores + causal mask ---
    scores_kernel<<<dim3(S_LEN / 64, S_LEN / 64), 256, 0, stream>>>(
        Qbf, Kbf, wmat, scores);

    // --- top-k ---
    topk_kernel<<<S_LEN, 256, 0, stream>>>(scores, topk);
}
